// KGCapsuleTransformer_81707457839516
// MI455X (gfx1250) — matrix operand order repaired and checked by reading, not compiled, not run on hardware
//
#include <hip/hip_runtime.h>

// Problem constants (match reference)
#define BSROWS 8192      // B*S
#define DDIM   1024      // D
#define NKNOW  20000     // N
#define TOPK   8
#define LN_EPS 1e-5f
#define SCORE_SCALE 0.03125f   // 1/sqrt(1024)

typedef __attribute__((ext_vector_type(8)))  __bf16 bf8v;
typedef __attribute__((ext_vector_type(16))) __bf16 bf16v;
typedef __attribute__((ext_vector_type(8)))  float  f8v;
typedef __attribute__((ext_vector_type(4)))  unsigned v4u;
typedef __attribute__((ext_vector_type(8)))  unsigned v8u;

static __device__ __forceinline__ unsigned short f2bf(float f) {
  unsigned u = __float_as_uint(f);
  unsigned r = 0x7FFFu + ((u >> 16) & 1u);      // round-to-nearest-even
  return (unsigned short)((u + r) >> 16);
}
static __device__ __forceinline__ float bf2f(unsigned short h) {
  return __uint_as_float(((unsigned)h) << 16);
}
static __device__ __forceinline__ bf8v as_bf8(uint4 v) {
  union { uint4 u; bf8v b; } x; x.u = v; return x.b;
}
static __device__ __forceinline__ bf16v combine16(uint4 lo, uint4 hi) {
  bf8v a = as_bf8(lo), b = as_bf8(hi);
  return __builtin_shufflevector(a, b, 0,1,2,3,4,5,6,7,8,9,10,11,12,13,14,15);
}

// LDS aperture puts the byte offset in addr[31:0] (ISA 10.2) — truncate flat ptr.
static __device__ __forceinline__ unsigned lds_off(const void* p) {
  return (unsigned)(uintptr_t)p;
}

// CDNA5 async global->LDS copy (ASYNCcnt-tracked, no VGPR data round trip).
static __device__ __forceinline__ void async_copy_b128(unsigned ldsOff, const void* g) {
  asm volatile("global_load_async_to_lds_b128 %0, %1, off"
               :: "v"(ldsOff), "v"(g) : "memory");
}
static __device__ __forceinline__ void wait_asynccnt0() {
  asm volatile("s_wait_asynccnt 0x0" ::: "memory");
}

// Tensor Data Mover: load one 16 x 2048B key tile (data_size=8B, tile 256x16)
// into LDS with 16B padding after every 1024B (pad_interval=7 -> 1024B,
// pad_amount=3 -> 4 DWORDs) so LDS rows have a 2080B stride.
// D# group0/group1 packed per ISA 8.3/8.4; VADDR2/3 NULL (2D tensor).
static __device__ __forceinline__ void tdm_load_keys(unsigned ldsOff, const void* g) {
  unsigned long long ga = (unsigned long long)(uintptr_t)g;
  v4u g0;
  g0[0] = 1u;                                   // count=1, user mode, no gather
  g0[1] = ldsOff;                               // lds_addr
  g0[2] = (unsigned)ga;                         // global_addr[31:0]
  g0[3] = ((unsigned)(ga >> 32) & 0x01FFFFFFu) | (2u << 30);  // addr hi | type=2
  const unsigned tensor_d0 = 256u, tensor_d1 = 16u;  // in 8B units
  const unsigned tile_d0 = 256u, tile_d1 = 16u;
  const unsigned stride0 = 256u;                // row stride, 8B units (=2048B)
  v8u g1;
  g1[0] = (3u << 16)      // data_size = 8B
        | (1u << 20)      // pad_enable
        | (7u << 22)      // pad_interval code 7 = 1024B
        | (3u << 25);     // pad_amount code 3 = 4 DWORDs = 16B
  g1[1] = (tensor_d0 & 0xFFFFu) << 16;                         // bits 63:48
  g1[2] = (tensor_d0 >> 16) | ((tensor_d1 & 0xFFFFu) << 16);   // 95:80
  g1[3] = (tensor_d1 >> 16) | (tile_d0 << 16);                 // 127:112
  g1[4] = tile_d1;                                             // 143:128 (tile_d2=0)
  g1[5] = stride0;                                             // tensor_dim0_stride lo
  g1[6] = 0u;
  g1[7] = 0u;
  asm volatile("tensor_load_to_lds %0, %1" :: "s"(g0), "s"(g1) : "memory");
}

// ---------------------------------------------------------------------------
// fp32 -> bf16 cast with strided destination (packs the [seq | agg] concat
// buffer for the K=2048 gate GEMM).
// ---------------------------------------------------------------------------
__global__ void cast_bf16_kernel(const float* __restrict__ src,
                                 unsigned short* __restrict__ dst,
                                 size_t rows, int cols, int dstStride) {
  size_t n = rows * (size_t)cols;
  for (size_t i = (size_t)blockIdx.x * blockDim.x + threadIdx.x; i < n;
       i += (size_t)gridDim.x * blockDim.x) {
    size_t r = i / (size_t)cols;
    int    c = (int)(i - r * (size_t)cols);
    dst[r * (size_t)dstStride + c] = f2bf(src[i]);
  }
}

__global__ void zero_f32_kernel(float* __restrict__ p, int n) {
  int i = blockIdx.x * blockDim.x + threadIdx.x;
  if (i < n) p[i] = 0.0f;
}

// ---------------------------------------------------------------------------
// Generic NT GEMM:  C[M,Ncols] = A[M,K](bf16,lda) * B[Ncols,K](bf16,ldb)^T + bias
// 64x64 block tile, 4 waves (wave32), each wave computes a 16x64 strip with
// v_wmma_f32_16x16x32_bf16.  Tiles staged with async global->LDS copies.
// ---------------------------------------------------------------------------
__global__ __launch_bounds__(128) void gemm_nt_bf16_kernel(
    const unsigned short* __restrict__ A, int lda,
    const unsigned short* __restrict__ Bm, int ldb,
    float* __restrict__ Cf, unsigned short* __restrict__ Cb, int ldc,
    const float* __restrict__ bias, int M, int Ncols, int Kt) {
  __shared__ __align__(16) unsigned short As[64][72];  // 144B row stride (16B mult)
  __shared__ __align__(16) unsigned short Bs[64][72];

  const int tid  = threadIdx.x;
  const int lane = tid & 31;
  const int wave = tid >> 5;
  const int half = lane >> 4;      // 0: lanes 0-15, 1: lanes 16-31
  const int mloc = lane & 15;

  const int rowBase = blockIdx.y * 64;
  const int colBase = blockIdx.x * 64;

  f8v acc[4];
#pragma unroll
  for (int t = 0; t < 4; ++t)
#pragma unroll
    for (int r = 0; r < 8; ++r) acc[t][r] = 0.0f;

  for (int kk = 0; kk < Kt; kk += 64) {
    // Stage A tile: each thread issues 4 async b128 copies for a 32-elem half row.
    {
      int r   = tid >> 1;
      int seg = (tid & 1) * 32;
      int ra  = rowBase + r; if (ra >= M) ra = M - 1;       // clamp (M=20000 case)
      const unsigned short* g = A + (size_t)ra * lda + kk + seg;
      unsigned lo = lds_off(&As[r][seg]);
#pragma unroll
      for (int i = 0; i < 4; ++i) async_copy_b128(lo + 16 * i, g + 8 * i);
    }
    // Stage B tile.
    {
      int r   = tid >> 1;
      int seg = (tid & 1) * 32;
      const unsigned short* g = Bm + (size_t)(colBase + r) * ldb + kk + seg;
      unsigned lo = lds_off(&Bs[r][seg]);
#pragma unroll
      for (int i = 0; i < 4; ++i) async_copy_b128(lo + 16 * i, g + 8 * i);
    }
    wait_asynccnt0();
    __syncthreads();

    const uint4* arow = (const uint4*)(&As[wave * 16 + mloc][0]);
#pragma unroll
    for (int kc = 0; kc < 64; kc += 32) {
      // A fragment (16x32 bf16): 16 contiguous K at kc+16h.
      uint4 alo = arow[(kc + 16 * half) >> 3];
      uint4 ahi = arow[((kc + 16 * half) >> 3) + 1];
      bf16v a = combine16(alo, ahi);
#pragma unroll
      for (int t = 0; t < 4; ++t) {
        // B fragment (32x16): lane holds col mloc; 16 contiguous K at kc+16h.
        const uint4* brow = (const uint4*)(&Bs[t * 16 + mloc][0]);
        uint4 blo = brow[(kc + 16 * half) >> 3];
        uint4 bhi = brow[((kc + 16 * half) >> 3) + 1];
        bf16v b = combine16(blo, bhi);
        acc[t] = __builtin_amdgcn_wmma_f32_16x16x32_bf16(
            false, a, false, b, (short)0, acc[t], false, false);
      }
    }
    __syncthreads();
  }

  // Store: C/D layout — VGPR r holds row r+8*half, col = lane&15 (per 16-tile).
  if (Cb) {
#pragma unroll
    for (int t = 0; t < 4; ++t) {
      int col = colBase + t * 16 + mloc;
#pragma unroll
      for (int r = 0; r < 8; ++r) {
        int row = rowBase + wave * 16 + r + 8 * half;
        if (row < M) Cb[(size_t)row * ldc + col] = f2bf(acc[t][r]);
      }
    }
  } else {
#pragma unroll
    for (int t = 0; t < 4; ++t) {
      int col = colBase + t * 16 + mloc;
      float bv = bias ? bias[col] : 0.0f;
#pragma unroll
      for (int r = 0; r < 8; ++r) {
        int row = rowBase + wave * 16 + r + 8 * half;
        if (row < M) Cf[(size_t)row * ldc + col] = acc[t][r] + bv;
      }
    }
  }
}

// ---------------------------------------------------------------------------
// Fused attention: per wave, 16 Q rows resident as 32 bf16 A-fragments
// (16x1024).  16-key chunks DMA'd into double-buffered LDS by the TDM
// (wave 0, TENSORcnt), 32 WMMAs/chunk, online softmax + register top-8 per
// row, then agg + usage scatter.
//
// LDS key row stride = 2080B (2048B data + 2x16B TDM pad); dynamic LDS (~73KB,
// within the 320KB/WGP budget).
// ---------------------------------------------------------------------------
#define KROWSTRIDE 2080
#define KBUFBYTES  (16 * KROWSTRIDE)   // 33280 per buffer

__global__ __launch_bounds__(128) void attn_topk_kernel(
    const unsigned short* __restrict__ qb,     // [BS, D] bf16
    const unsigned short* __restrict__ keysb,  // [N, D] bf16
    const unsigned short* __restrict__ valsb,  // [N, D] bf16
    float* __restrict__ aggf,                  // [BS, D] fp32
    unsigned short* __restrict__ catb,         // [BS, 2D] bf16 (write agg at +D)
    float* __restrict__ usage) {               // [N]
  extern __shared__ __align__(16) unsigned char smem[];
  unsigned char* ksBase = smem;                         // 2 x KBUFBYTES
  float* scoreS = (float*)(smem + 2 * KBUFBYTES);       // [4][16][16]
  float* topWS  = scoreS + 4 * 16 * 16;                 // [4][16][8]
  int*   topIS  = (int*)(topWS + 4 * 16 * TOPK);        // [4][16][8]

  const int tid  = threadIdx.x;
  const int lane = tid & 31;
  const int wave = tid >> 5;
  const int half = lane >> 4;
  const int mloc = lane & 15;
  const int rowBase = blockIdx.x * 64 + wave * 16;

  // Resident Q fragments for this wave's 16 rows (32 chunks of K=32).
  const int grow = rowBase + mloc;
  const uint4* qrow = (const uint4*)(qb + (size_t)grow * DDIM);
  bf16v qf[32];
#pragma unroll
  for (int c = 0; c < 32; ++c)
    qf[c] = combine16(qrow[4 * c + 2 * half], qrow[4 * c + 2 * half + 1]);

  // Online softmax + top-8 state (meaningful in lanes 0..15; lane = local row).
  float m = -1e30f, z = 0.0f;
  float ts[TOPK];
  int   ti[TOPK];
#pragma unroll
  for (int k = 0; k < TOPK; ++k) { ts[k] = -1e30f; ti[k] = 0; }

  // Prologue: DMA chunk 0 into buffer 0.
  if (wave == 0) tdm_load_keys(lds_off(ksBase), keysb);

  for (int nb = 0; nb < NKNOW; nb += 16) {
    const int buf = (nb >> 4) & 1;
    __builtin_amdgcn_s_wait_tensorcnt(0);   // issuing wave drains its TDM
    __syncthreads();
    // Kick DMA of the next chunk into the other buffer (overlaps compute).
    if (wave == 0 && nb + 16 < NKNOW)
      tdm_load_keys(lds_off(ksBase + (buf ^ 1) * KBUFBYTES),
                    keysb + (size_t)(nb + 16) * DDIM);

    // 16x16 score tile: 32 WMMAs over K=1024.
    f8v acc;
#pragma unroll
    for (int r = 0; r < 8; ++r) acc[r] = 0.0f;
    const uint4* brow =
        (const uint4*)(ksBase + buf * KBUFBYTES + mloc * KROWSTRIDE);
#pragma unroll
    for (int c = 0; c < 32; ++c) {
      // chunk c = bytes [64c,64c+64); TDM pad adds 16B after the first 1024B.
      int bidx = 4 * c + 2 * half + (c >= 16 ? 1 : 0);
      uint4 blo = brow[bidx];
      uint4 bhi = brow[bidx + 1];
      bf16v b = combine16(blo, bhi);
      acc = __builtin_amdgcn_wmma_f32_16x16x32_bf16(
          false, qf[c], false, b, (short)0, acc, false, false);
    }
#pragma unroll
    for (int r = 0; r < 8; ++r)
      scoreS[(wave * 16 + r + 8 * half) * 16 + mloc] = acc[r] * SCORE_SCALE;
    __syncthreads();

    // Lanes 0..15: scan this chunk's 16 scores for their row.
    if (lane < 16) {
      for (int j = 0; j < 16; ++j) {
        float s = scoreS[(wave * 16 + lane) * 16 + j];
        float nm = fmaxf(m, s);
        z = z * __expf(m - nm) + __expf(s - nm);
        m = nm;
        int amin = 0; float vmin = ts[0];
#pragma unroll
        for (int k = 1; k < TOPK; ++k)
          if (ts[k] < vmin) { vmin = ts[k]; amin = k; }
        bool ins = s > vmin;
#pragma unroll
        for (int k = 0; k < TOPK; ++k)
          if (ins && k == amin) { ts[k] = s; ti[k] = nb + j; }
      }
    }
  }

  __syncthreads();
  // Finalize attn weights for the top-8; scatter usage counts.
  if (lane < 16) {
    float inv = 1.0f / z;
#pragma unroll
    for (int k = 0; k < TOPK; ++k) {
      float w = __expf(ts[k] - m) * inv;
      topWS[(wave * 16 + lane) * TOPK + k] = w;
      topIS[(wave * 16 + lane) * TOPK + k] = ti[k];
      atomicAdd(&usage[ti[k]], w);
    }
  }
  __syncthreads();

  // agg[row,:] = sum_k w_k * vals[idx_k,:]; also write bf16 into concat buffer.
  for (int r = 0; r < 16; ++r) {
    int row = rowBase + r;
    float w[TOPK]; int idx[TOPK];
#pragma unroll
    for (int k = 0; k < TOPK; ++k) {
      w[k]   = topWS[(wave * 16 + r) * TOPK + k];
      idx[k] = topIS[(wave * 16 + r) * TOPK + k];
    }
    for (int d = lane; d < DDIM; d += 32) {
      float a = 0.0f;
#pragma unroll
      for (int k = 0; k < TOPK; ++k)
        a += w[k] * bf2f(valsb[(size_t)idx[k] * DDIM + d]);
      aggf[(size_t)row * DDIM + d] = a;
      catb[(size_t)row * (2 * DDIM) + DDIM + d] = f2bf(a);
    }
  }
}

// ---------------------------------------------------------------------------
// enhanced = sigmoid(gpre)*agg + (1-sigmoid(gpre))*seq, stored bf16.
// ---------------------------------------------------------------------------
__global__ void gate_kernel(const float* __restrict__ gpre,
                            const float* __restrict__ seq,
                            const float* __restrict__ agg,
                            unsigned short* __restrict__ enhb, size_t n) {
  for (size_t i = (size_t)blockIdx.x * blockDim.x + threadIdx.x; i < n;
       i += (size_t)gridDim.x * blockDim.x) {
    float g = 1.0f / (1.0f + __expf(-gpre[i]));
    enhb[i] = f2bf(g * agg[i] + (1.0f - g) * seq[i]);
  }
}

// ---------------------------------------------------------------------------
// LayerNorm(h) * g + b + seq  (one 256-thread block per row of 1024).
// ---------------------------------------------------------------------------
__global__ __launch_bounds__(256) void ln_residual_kernel(
    const float* __restrict__ h, const float* __restrict__ seq,
    const float* __restrict__ g, const float* __restrict__ b,
    float* __restrict__ out) {
  __shared__ float red[256];
  const int row = blockIdx.x;
  const int t = threadIdx.x;
  const float* hr = h + (size_t)row * DDIM;
  float v[4]; float s = 0.0f;
#pragma unroll
  for (int i = 0; i < 4; ++i) { v[i] = hr[t + 256 * i]; s += v[i]; }
  red[t] = s; __syncthreads();
  for (int o = 128; o > 0; o >>= 1) { if (t < o) red[t] += red[t + o]; __syncthreads(); }
  float mu = red[0] * (1.0f / DDIM);
  __syncthreads();
  float s2 = 0.0f;
#pragma unroll
  for (int i = 0; i < 4; ++i) { float d = v[i] - mu; s2 += d * d; }
  red[t] = s2; __syncthreads();
  for (int o = 128; o > 0; o >>= 1) { if (t < o) red[t] += red[t + o]; __syncthreads(); }
  float rstd = rsqrtf(red[0] * (1.0f / DDIM) + LN_EPS);
#pragma unroll
  for (int i = 0; i < 4; ++i) {
    int col = t + 256 * i;
    out[(size_t)row * DDIM + col] =
        (v[i] - mu) * rstd * g[col] + b[col] + seq[(size_t)row * DDIM + col];
  }
}

// ---------------------------------------------------------------------------
extern "C" void kernel_launch(void* const* d_in, const int* in_sizes, int n_in,
                              void* d_out, int out_size, void* d_ws, size_t ws_size,
                              hipStream_t stream) {
  (void)in_sizes; (void)n_in; (void)out_size; (void)ws_size;
  const float* seq = (const float*)d_in[0];
  const float* ke  = (const float*)d_in[1];
  const float* Wq  = (const float*)d_in[2];
  const float* Wk  = (const float*)d_in[3];
  const float* Wv  = (const float*)d_in[4];
  const float* Wg  = (const float*)d_in[5];
  const float* bg  = (const float*)d_in[6];
  const float* Wo  = (const float*)d_in[7];
  const float* bo  = (const float*)d_in[8];
  const float* lng = (const float*)d_in[9];
  const float* lnb = (const float*)d_in[10];

  float* out   = (float*)d_out;
  float* usage = out + (size_t)BSROWS * DDIM;

  char* ws = (char*)d_ws;
  size_t off = 0;
  auto alloc = [&](size_t bytes) -> void* {
    void* p = ws + off;
    off = (off + bytes + 255) & ~(size_t)255;
    return p;
  };
  unsigned short* catb  = (unsigned short*)alloc((size_t)BSROWS * 2 * DDIM * 2); // [BS,2D]
  unsigned short* keb   = (unsigned short*)alloc((size_t)NKNOW * DDIM * 2);
  unsigned short* wqb   = (unsigned short*)alloc((size_t)DDIM * DDIM * 2);
  unsigned short* wkb   = (unsigned short*)alloc((size_t)DDIM * DDIM * 2);
  unsigned short* wvb   = (unsigned short*)alloc((size_t)DDIM * DDIM * 2);
  unsigned short* wgb   = (unsigned short*)alloc((size_t)DDIM * 2 * DDIM * 2);   // [D,2D]
  unsigned short* wob   = (unsigned short*)alloc((size_t)DDIM * DDIM * 2);
  unsigned short* qb    = (unsigned short*)alloc((size_t)BSROWS * DDIM * 2);
  unsigned short* keysb = (unsigned short*)alloc((size_t)NKNOW * DDIM * 2);
  unsigned short* valsb = (unsigned short*)alloc((size_t)NKNOW * DDIM * 2);
  unsigned short* enhb  = (unsigned short*)alloc((size_t)BSROWS * DDIM * 2);
  float* aggf = (float*)alloc((size_t)BSROWS * DDIM * 4);
  float* gpre = (float*)alloc((size_t)BSROWS * DDIM * 4);  // reused as h later

  // --- Phase 0: casts + zero usage ------------------------------------------
  cast_bf16_kernel<<<dim3(4096), dim3(256), 0, stream>>>(seq, catb, BSROWS, DDIM, 2 * DDIM);
  cast_bf16_kernel<<<dim3(8192), dim3(256), 0, stream>>>(ke, keb, NKNOW, DDIM, DDIM);
  cast_bf16_kernel<<<dim3(1024), dim3(256), 0, stream>>>(Wq, wqb, DDIM, DDIM, DDIM);
  cast_bf16_kernel<<<dim3(1024), dim3(256), 0, stream>>>(Wk, wkb, DDIM, DDIM, DDIM);
  cast_bf16_kernel<<<dim3(1024), dim3(256), 0, stream>>>(Wv, wvb, DDIM, DDIM, DDIM);
  cast_bf16_kernel<<<dim3(2048), dim3(256), 0, stream>>>(Wg, wgb, DDIM, 2 * DDIM, 2 * DDIM);
  cast_bf16_kernel<<<dim3(1024), dim3(256), 0, stream>>>(Wo, wob, DDIM, DDIM, DDIM);
  zero_f32_kernel<<<dim3((NKNOW + 255) / 256), dim3(256), 0, stream>>>(usage, NKNOW);

  // --- Phase 1: projections (WMMA bf16) -------------------------------------
  gemm_nt_bf16_kernel<<<dim3(DDIM / 64, BSROWS / 64), dim3(128), 0, stream>>>(
      catb, 2 * DDIM, wqb, DDIM, nullptr, qb, DDIM, nullptr, BSROWS, DDIM, DDIM);
  gemm_nt_bf16_kernel<<<dim3(DDIM / 64, (NKNOW + 63) / 64), dim3(128), 0, stream>>>(
      keb, DDIM, wkb, DDIM, nullptr, keysb, DDIM, nullptr, NKNOW, DDIM, DDIM);
  gemm_nt_bf16_kernel<<<dim3(DDIM / 64, (NKNOW + 63) / 64), dim3(128), 0, stream>>>(
      keb, DDIM, wvb, DDIM, nullptr, valsb, DDIM, nullptr, NKNOW, DDIM, DDIM);

  // --- Phase 2: fused scores/softmax/top-k/agg/usage (TDM-fed) --------------
  const int attnLds = 2 * KBUFBYTES + (4 * 16 * 16) * 4 + 2 * (4 * 16 * TOPK) * 4;
  (void)hipFuncSetAttribute((const void*)attn_topk_kernel,
                            hipFuncAttributeMaxDynamicSharedMemorySize, attnLds);
  attn_topk_kernel<<<dim3(BSROWS / 64), dim3(128), attnLds, stream>>>(
      qb, keysb, valsb, aggf, catb, usage);

  // --- Phase 3: gate GEMM (K=2048 over [seq|agg]) + gating ------------------
  gemm_nt_bf16_kernel<<<dim3(DDIM / 64, BSROWS / 64), dim3(128), 0, stream>>>(
      catb, 2 * DDIM, wgb, 2 * DDIM, gpre, nullptr, DDIM, bg, BSROWS, DDIM, 2 * DDIM);
  gate_kernel<<<dim3(4096), dim3(256), 0, stream>>>(
      gpre, seq, aggf, enhb, (size_t)BSROWS * DDIM);

  // --- Phase 4: output projection (h reuses gpre buffer) + LN + residual ----
  gemm_nt_bf16_kernel<<<dim3(DDIM / 64, BSROWS / 64), dim3(128), 0, stream>>>(
      enhb, DDIM, wob, DDIM, gpre, nullptr, DDIM, bo, BSROWS, DDIM, DDIM);
  ln_residual_kernel<<<dim3(BSROWS), dim3(256), 0, stream>>>(gpre, seq, lng, lnb, out);
}
